// SurrogateGAT_85985245266466
// MI455X (gfx1250) — compile-verified, matrix operand order checked by reading, not run on hardware
//
#include <hip/hip_runtime.h>
#include <hip/hip_bf16.h>

#define NN    50000
#define ERAW  800000
#define ETOT  850000
#define SLOPE 0.2f

typedef __attribute__((ext_vector_type(16))) _Float16 v16h;
typedef __attribute__((ext_vector_type(8)))  float    v8f;

// monotonic f32 -> u32 mapping for atomicMax-based segment max
__device__ inline unsigned fmap(float f) {
  unsigned u = __builtin_bit_cast(unsigned, f);
  return (u & 0x80000000u) ? ~u : (u | 0x80000000u);
}
__device__ inline float funmap(unsigned u) {
  unsigned v = (u & 0x80000000u) ? (u & 0x7FFFFFFFu) : ~u;
  return __builtin_bit_cast(float, v);
}
#define MINIT 0x007FFFFFu   // fmap(-inf)

__device__ inline void edge_ends(const int* __restrict__ ei, int e, int& s, int& d) {
  if (e < ERAW) { s = ei[e]; d = ei[ERAW + e]; } else { s = d = e - ERAW; }
}

// ---------------- GEMM: O[M,128] = A[M,128] @ W[128,128], f16 WMMA ----------------
__global__ void __launch_bounds__(256) gemm128(const float* __restrict__ A,
                                               const float* __restrict__ W,
                                               float* __restrict__ O, int mTiles) {
  __shared__ _Float16 Wt[128 * 128];          // W transposed [n][k], 32 KB
  const int tid = threadIdx.x;
  for (int i = tid; i < 128 * 128; i += 256) {
    int k = i >> 7, n = i & 127;
    Wt[n * 128 + k] = (_Float16)W[i];         // native v_cvt_f16_f32
  }
  __syncthreads();
  const int wave = tid >> 5, lane = tid & 31;
  const int mTile = blockIdx.x * 8 + wave;
  if (mTile >= mTiles) return;
  const int rlo = lane & 15;
  const int kHalf = (lane >> 4) << 3;         // 0 or 8
  const float* arow = A + (size_t)(mTile * 16 + rlo) * 128;
  v8f acc[8] = {};
  for (int k0 = 0; k0 < 128; k0 += 32) {
    v16h af;
    const float* ap = arow + k0 + kHalf;
#pragma unroll
    for (int i = 0; i < 8; ++i) af[i] = (_Float16)ap[i];
#pragma unroll
    for (int i = 0; i < 8; ++i) af[8 + i] = (_Float16)ap[16 + i];
#pragma unroll
    for (int nt = 0; nt < 8; ++nt) {
      const _Float16* bp = &Wt[(nt * 16 + rlo) * 128 + k0 + kHalf];
      v16h bf;
#pragma unroll
      for (int i = 0; i < 8; ++i) bf[i] = bp[i];
#pragma unroll
      for (int i = 0; i < 8; ++i) bf[8 + i] = bp[16 + i];
      acc[nt] = __builtin_amdgcn_wmma_f32_16x16x32_f16(
          false, af, false, bf, (short)0, acc[nt], false, false);
    }
  }
  const int rbase = mTile * 16 + ((lane >> 4) << 3);
#pragma unroll
  for (int nt = 0; nt < 8; ++nt)
#pragma unroll
    for (int v = 0; v < 8; ++v)
      O[(size_t)(rbase + v) * 128 + nt * 16 + rlo] = acc[nt][v];
}

// ---------------- logits: O[M,40] = [A1|A2][M,256] @ Wl[256,40] + bl ----------------
__global__ void __launch_bounds__(256) gemm_cat(const float* __restrict__ A1,
                                                const float* __restrict__ A2,
                                                const float* __restrict__ Wl,
                                                const float* __restrict__ bl,
                                                float* __restrict__ O, int mTiles) {
  __shared__ _Float16 Wt[48 * 256];           // padded-N transposed, 24 KB
  const int tid = threadIdx.x;
  for (int i = tid; i < 48 * 256; i += 256) {
    int n = i >> 8, k = i & 255;
    Wt[i] = (n < 40) ? (_Float16)Wl[k * 40 + n] : (_Float16)0.0f;
  }
  __syncthreads();
  const int wave = tid >> 5, lane = tid & 31;
  const int mTile = blockIdx.x * 8 + wave;
  if (mTile >= mTiles) return;
  const int rlo = lane & 15;
  const int kHalf = (lane >> 4) << 3;
  const int row = mTile * 16 + rlo;
  v8f acc[3] = {};
  for (int k0 = 0; k0 < 256; k0 += 32) {
    const float* base = (k0 < 128) ? (A1 + (size_t)row * 128 + k0)
                                   : (A2 + (size_t)row * 128 + (k0 - 128));
    const float* ap = base + kHalf;
    v16h af;
#pragma unroll
    for (int i = 0; i < 8; ++i) af[i] = (_Float16)ap[i];
#pragma unroll
    for (int i = 0; i < 8; ++i) af[8 + i] = (_Float16)ap[16 + i];
#pragma unroll
    for (int nt = 0; nt < 3; ++nt) {
      const _Float16* bp = &Wt[(nt * 16 + rlo) * 256 + k0 + kHalf];
      v16h bf;
#pragma unroll
      for (int i = 0; i < 8; ++i) bf[i] = bp[i];
#pragma unroll
      for (int i = 0; i < 8; ++i) bf[8 + i] = bp[16 + i];
      acc[nt] = __builtin_amdgcn_wmma_f32_16x16x32_f16(
          false, af, false, bf, (short)0, acc[nt], false, false);
    }
  }
  const int rbase = mTile * 16 + ((lane >> 4) << 3);
#pragma unroll
  for (int nt = 0; nt < 3; ++nt) {
    const int col = nt * 16 + rlo;
    if (col < 40) {
      float bb = bl[col];
#pragma unroll
      for (int v = 0; v < 8; ++v)
        O[(size_t)(rbase + v) * 40 + col] = acc[nt][v] + bb;
    }
  }
}

// ---------------- attention logits per node ----------------
__global__ void attn_logits(const float* __restrict__ h, const float* __restrict__ asrc,
                            const float* __restrict__ adst, float* __restrict__ es,
                            float* __restrict__ ed) {
  int t = blockIdx.x * blockDim.x + threadIdx.x;
  if (t >= NN * 2) return;
  int n = t >> 1, hd = t & 1;
  const float* hp = h + (size_t)n * 128 + hd * 64;
  const float* as = asrc + hd * 64;
  const float* ad = adst + hd * 64;
  float s0 = 0.f, s1 = 0.f;
  for (int c = 0; c < 64; ++c) { float v = hp[c]; s0 += v * as[c]; s1 += v * ad[c]; }
  es[t] = s0; ed[t] = s1;
}

// ---------------- per-layer init: zero accumulator + denom, seed max ----------------
__global__ void init_layer(float* __restrict__ acc, float* __restrict__ denom,
                           unsigned* __restrict__ mu) {
  int i = blockIdx.x * blockDim.x + threadIdx.x;
  if (i < NN * 128) acc[i] = 0.f;
  if (i < NN * 2) { denom[i] = 0.f; mu[i] = MINIT; }
}

__global__ void edge_max(const int* __restrict__ ei, const float* __restrict__ es,
                         const float* __restrict__ ed, unsigned* __restrict__ mu) {
  int t = blockIdx.x * blockDim.x + threadIdx.x;
  if (t >= ETOT * 2) return;
  int e = t >> 1, hd = t & 1, s, d;
  edge_ends(ei, e, s, d);
  float v = es[s * 2 + hd] + ed[d * 2 + hd];
  v = (v > 0.f) ? v : SLOPE * v;
  atomicMax(&mu[d * 2 + hd], fmap(v));
}

__global__ void fix_m(unsigned* __restrict__ mu) {
  int i = blockIdx.x * blockDim.x + threadIdx.x;
  if (i >= NN * 2) return;
  unsigned u = mu[i];
  float v = (u == MINIT) ? 0.f : funmap(u);   // where(isneginf(m), 0, m)
  ((float*)mu)[i] = v;
}

__global__ void edge_w(const int* __restrict__ ei, const float* __restrict__ es,
                       const float* __restrict__ ed, const float* __restrict__ m,
                       float* __restrict__ wbuf, float* __restrict__ denom) {
  int t = blockIdx.x * blockDim.x + threadIdx.x;
  if (t >= ETOT * 2) return;
  int e = t >> 1, hd = t & 1, s, d;
  edge_ends(ei, e, s, d);
  float v = es[s * 2 + hd] + ed[d * 2 + hd];
  v = (v > 0.f) ? v : SLOPE * v;
  float w = __expf(v - m[d * 2 + hd]);
  wbuf[t] = w;
  __hip_atomic_fetch_add(&denom[d * 2 + hd], w, __ATOMIC_RELAXED, __HIP_MEMORY_SCOPE_AGENT);
}

__global__ void edge_scatter(const int* __restrict__ ei, const float* __restrict__ wbuf,
                             const float* __restrict__ denom, const float* __restrict__ h,
                             float* __restrict__ acc) {
  int t = blockIdx.x * blockDim.x + threadIdx.x;
  if (t >= ETOT * 128) return;
  int e = t >> 7, c = t & 127, hd = c >> 6, s, d;
  edge_ends(ei, e, s, d);
  float alpha = wbuf[e * 2 + hd] / fmaxf(denom[d * 2 + hd], 1e-16f);
  float val = alpha * h[(size_t)s * 128 + c];
  __hip_atomic_fetch_add(&acc[(size_t)d * 128 + c], val, __ATOMIC_RELAXED,
                         __HIP_MEMORY_SCOPE_AGENT);
}

__global__ void finish_elu(float* __restrict__ hbuf, const float* __restrict__ b) {
  int i = blockIdx.x * blockDim.x + threadIdx.x;
  if (i >= NN * 128) return;
  float v = hbuf[i] + b[i & 127];
  hbuf[i] = (v > 0.f) ? v : (__expf(v) - 1.f);
}

__global__ void logsoftmax40(const float* __restrict__ logits, float* __restrict__ out) {
  int n = blockIdx.x * blockDim.x + threadIdx.x;
  if (n >= NN) return;
  const float* lp = logits + (size_t)n * 40;
  float mx = lp[0];
  for (int c = 1; c < 40; ++c) mx = fmaxf(mx, lp[c]);
  float s = 0.f;
  for (int c = 0; c < 40; ++c) s += __expf(lp[c] - mx);
  float lse = mx + __logf(s);
  for (int c = 0; c < 40; ++c) out[(size_t)n * 40 + c] = lp[c] - lse;
}

extern "C" void kernel_launch(void* const* d_in, const int* in_sizes, int n_in,
                              void* d_out, int out_size, void* d_ws, size_t ws_size,
                              hipStream_t stream) {
  const float* x     = (const float*)d_in[0];
  const int*   ei    = (const int*)  d_in[1];
  const float* W0    = (const float*)d_in[2];
  const float* asrc0 = (const float*)d_in[3];
  const float* adst0 = (const float*)d_in[4];
  const float* b0    = (const float*)d_in[5];
  const float* W1    = (const float*)d_in[6];
  const float* asrc1 = (const float*)d_in[7];
  const float* adst1 = (const float*)d_in[8];
  const float* b1    = (const float*)d_in[9];
  const float* linw  = (const float*)d_in[10];
  const float* linb  = (const float*)d_in[11];
  float* out = (float*)d_out;

  float* ws = (float*)d_ws;
  float* h0     = ws;                    // [N,128] shared GEMM output (per layer)
  float* h1     = h0 + (size_t)NN * 128; // [N,128] layer-0 result (scatter target)
  float* h2     = h1 + (size_t)NN * 128; // [N,128] layer-1 result (scatter target)
  float* es     = h2 + (size_t)NN * 128; // [N,2]
  float* ed     = es + NN * 2;           // [N,2]
  float* mbuf   = ed + NN * 2;           // [N,2] (u32-mapped max, then float)
  float* denom  = mbuf + NN * 2;         // [N,2]
  float* wbuf   = denom + NN * 2;        // [ETOT,2]
  float* logits = wbuf + (size_t)ETOT * 2; // [N,40]

  const int mTiles = NN / 16;                       // 3125
  dim3 blk(256);
  dim3 gGemm((mTiles + 7) / 8);                     // 8 waves / block
  dim3 gN128((NN * 128 + 255) / 256);
  dim3 gN2((NN * 2 + 255) / 256);
  dim3 gE2((ETOT * 2 + 255) / 256);
  dim3 gEc((ETOT * 128 + 255) / 256);
  dim3 gN((NN + 255) / 256);

  // ---- layer 0 ----
  init_layer<<<gN128, blk, 0, stream>>>(h1, denom, (unsigned*)mbuf);
  gemm128<<<gGemm, blk, 0, stream>>>(x, W0, h0, mTiles);
  attn_logits<<<gN2, blk, 0, stream>>>(h0, asrc0, adst0, es, ed);
  edge_max<<<gE2, blk, 0, stream>>>(ei, es, ed, (unsigned*)mbuf);
  fix_m<<<gN2, blk, 0, stream>>>((unsigned*)mbuf);
  edge_w<<<gE2, blk, 0, stream>>>(ei, es, ed, mbuf, wbuf, denom);
  edge_scatter<<<gEc, blk, 0, stream>>>(ei, wbuf, denom, h0, h1);
  finish_elu<<<gN128, blk, 0, stream>>>(h1, b0);

  // ---- layer 1 ----
  init_layer<<<gN128, blk, 0, stream>>>(h2, denom, (unsigned*)mbuf);
  gemm128<<<gGemm, blk, 0, stream>>>(h1, W1, h0, mTiles);
  attn_logits<<<gN2, blk, 0, stream>>>(h0, asrc1, adst1, es, ed);
  edge_max<<<gE2, blk, 0, stream>>>(ei, es, ed, (unsigned*)mbuf);
  fix_m<<<gN2, blk, 0, stream>>>((unsigned*)mbuf);
  edge_w<<<gE2, blk, 0, stream>>>(ei, es, ed, mbuf, wbuf, denom);
  edge_scatter<<<gEc, blk, 0, stream>>>(ei, wbuf, denom, h0, h2);
  finish_elu<<<gN128, blk, 0, stream>>>(h2, b1);

  // ---- head ----
  gemm_cat<<<gGemm, blk, 0, stream>>>(h1, h2, linw, linb, logits, mTiles);
  logsoftmax40<<<gN, blk, 0, stream>>>(logits, out);
}
